// FWPBlock_36017595744387
// MI455X (gfx1250) — matrix operand order repaired and checked by reading, not compiled, not run on hardware
//
#include <hip/hip_runtime.h>

// ---------- CDNA5 types / WMMA ----------
typedef __bf16 bf16_t;
typedef __attribute__((ext_vector_type(16))) __bf16 v16bf;
typedef __attribute__((ext_vector_type(8)))  __bf16 v8bf;
typedef __attribute__((ext_vector_type(8)))  float  v8f;
typedef __attribute__((ext_vector_type(4)))  float  v4f;   // clang vector: valid for nontemporal builtins

static __device__ __forceinline__ bf16_t f2bf(float f) { return (bf16_t)f; }

#define WMMA_BF16(a, b, c) \
  __builtin_amdgcn_wmma_f32_16x16x32_bf16(false, (a), false, (b), (short)0, (c), false, false)

#define CAT16(lo, hi) __builtin_shufflevector((lo), (hi), 0, 1, 2, 3, 4, 5, 6, 7, 8, 9, 10, 11, 12, 13, 14, 15)

#if defined(__HIP_DEVICE_COMPILE__) && __has_builtin(__builtin_amdgcn_global_load_async_to_lds_b128) && __has_builtin(__builtin_amdgcn_s_wait_asynccnt)
#define HAVE_ASYNC_LDS 1
// exact param types per hipcc diagnostic: int __vector(4)* with AS1 (global) / AS3 (LDS)
typedef int v4i_t __attribute__((vector_size(16)));
typedef __attribute__((address_space(1))) v4i_t* gv4p;
typedef __attribute__((address_space(3))) v4i_t* lv4p;
#endif

// ---------- problem constants (from reference setup_inputs) ----------
static constexpr int B_    = 8;
static constexpr int T_    = 1024;
static constexpr int F_    = 512;
static constexpr int D_    = 64;
static constexpr int NROW  = B_ * T_;      // 8192
static constexpr int CHUNK = 64;
static constexpr int NCH   = T_ / CHUNK;   // 16

// =====================================================================
// Kernel 0: convert+transpose weights -> bf16 Wt[w][d][f] (B-fragment
// elements become 16 consecutive bf16 = one 32B vector load).
// =====================================================================
__global__ __launch_bounds__(256)
void w_prep(const float* __restrict__ Wk, const float* __restrict__ Wq,
            const float* __restrict__ Wv, bf16_t* __restrict__ Wt) {
  const int idx = blockIdx.x * 256 + threadIdx.x;      // over 3*F*D
  if (idx >= 3 * F_ * D_) return;
  const int w = idx / (F_ * D_), rem = idx % (F_ * D_);
  const int f = rem / D_, d = rem % D_;
  const float* W = (w == 0) ? Wk : (w == 1) ? Wq : Wv;
  Wt[(w * D_ + d) * F_ + f] = f2bf(W[rem]);
}

// =====================================================================
// Kernel A: LayerNorm + 3 projections (bf16 WMMA) + relu + sum-normalize
// 1 wave per 16-row tile. Grid = NROW/16 = 512 blocks of 32 threads.
// =====================================================================
__global__ __launch_bounds__(32)
void proj_ln_gemm(const float* __restrict__ x,
                  const bf16_t* __restrict__ Wt,     // [3][D][F] bf16
                  const float* __restrict__ gamma,
                  const float* __restrict__ beta,
                  float* __restrict__ Kout,
                  float* __restrict__ Qout,
                  float* __restrict__ Vout) {
  __shared__ bf16_t xn[16][F_];        // 16 KB: layernormed rows in bf16
  __shared__ float  outt[3][16][D_];   // 12 KB: raw K/Q/V tiles before norm
  const int  lane = threadIdx.x;       // wave32
  const long row0 = (long)blockIdx.x * 16;

  // ---- LayerNorm 16 rows (f32 stats, wave32 shfl reductions) ----
  for (int r = 0; r < 16; ++r) {
    const float* xr = x + (row0 + r) * F_;
    float s = 0.f, ss = 0.f;
    for (int c = lane; c < F_; c += 32) { float v = xr[c]; s += v; ss += v * v; }
    #pragma unroll
    for (int m = 16; m >= 1; m >>= 1) { s += __shfl_xor(s, m, 32); ss += __shfl_xor(ss, m, 32); }
    const float mu   = s * (1.0f / F_);
    const float var  = ss * (1.0f / F_) - mu * mu;
    const float rstd = rsqrtf(var + 1e-5f);
    for (int c = lane; c < F_; c += 32)
      xn[r][c] = f2bf((xr[c] - mu) * rstd * gamma[c] + beta[c]);
  }
  __syncthreads();

  // ---- GEMM: [16 x 512] (bf16, LDS) x [512 x 64] (bf16, global) ----
  // A fragment (16-bit A 16x32): lane half h, row m:
//   elements 0..7  = K (16h   .. 16h+7)  -> contiguous 16B
//   elements 8..15 = K (16h+8 .. 16h+15) -> contiguous 16B
  // B fragment (16-bit B 32x16): lane half h, col:
  //   elements 0..15 = K (16h .. 16h+15)  -> contiguous 32B in Wt[w][col][*]
  const int m = lane & 15, h = lane >> 4;
  #pragma unroll
  for (int w = 0; w < 3; ++w) {
    for (int n = 0; n < 4; ++n) {           // 4 column tiles of 16 -> D=64
      const int col = n * 16 + m;
      const bf16_t* Wrow = Wt + (long)(w * D_ + col) * F_;
      v8f acc = {};
      for (int kk = 0; kk < F_; kk += 32) { // 16 WMMAs per tile
        const v8bf alo = *(const v8bf*)&xn[m][kk + 16 * h];
        const v8bf ahi = *(const v8bf*)&xn[m][kk + 16 * h + 8];
        const v16bf a  = CAT16(alo, ahi);
        const v16bf b  = *(const v16bf*)&Wrow[kk + 16 * h];
        acc = WMMA_BF16(a, b, acc);
      }
      #pragma unroll
      for (int r = 0; r < 8; ++r)           // C layout: lane half h -> rows r+8h
        outt[w][r + 8 * h][n * 16 + m] = acc[r];
    }
  }
  __syncthreads();

  // ---- relu + sum-normalize K,Q ; V passthrough ----
  for (int r = 0; r < 16; ++r) {
    float k0 = fmaxf(outt[0][r][lane], 0.f), k1 = fmaxf(outt[0][r][lane + 32], 0.f);
    float q0 = fmaxf(outt[1][r][lane], 0.f), q1 = fmaxf(outt[1][r][lane + 32], 0.f);
    float ks = k0 + k1, qs = q0 + q1;
    #pragma unroll
    for (int mm = 16; mm >= 1; mm >>= 1) { ks += __shfl_xor(ks, mm, 32); qs += __shfl_xor(qs, mm, 32); }
    const float kinv = 1.0f / (1e-5f + ks);
    const float qinv = 1.0f / (1e-5f + qs);
    const long o = (row0 + r) * D_;
    Kout[o + lane] = k0 * kinv;  Kout[o + lane + 32] = k1 * kinv;
    Qout[o + lane] = q0 * qinv;  Qout[o + lane + 32] = q1 * qinv;
    Vout[o + lane] = outt[2][r][lane];  Vout[o + lane + 32] = outt[2][r][lane + 32];
  }
}

// =====================================================================
// Kernel B: per-chunk Sum_t V[t] (x) K[t]  ==  V_chunk^T @ K_chunk
// 64x64x64 GEMM per (b,c) via bf16 WMMA, transposed LDS staging so all
// fragments are contiguous b128 LDS loads. Grid = 128 blocks of 128.
// =====================================================================
__global__ __launch_bounds__(128)
void chunk_outer_sum(const float* __restrict__ Kw,
                     const float* __restrict__ Vw,
                     float* __restrict__ csum) {
  __shared__ bf16_t KcT[D_][CHUNK];  // KcT[j][t] : 8 KB
  __shared__ bf16_t VcT[D_][CHUNK];  // VcT[i][t] : 8 KB
  const int  tid  = threadIdx.x;
  const int  bc   = blockIdx.x;              // b*NCH + c ; global row base = bc*64
  const long base = (long)bc * CHUNK * D_;
  for (int idx = tid; idx < CHUNK * D_; idx += 128) {
    const int t = idx / D_, d = idx % D_;
    KcT[d][t] = f2bf(Kw[base + idx]);
    VcT[d][t] = f2bf(Vw[base + idx]);
  }
  __syncthreads();

  const int lane = tid & 31, wave = tid >> 5;   // 4 waves: row bands of 16
  const int m = lane & 15, h = lane >> 4;
  const int irow = wave * 16 + m;               // A row index i
  for (int n = 0; n < 4; ++n) {
    const int col = n * 16 + m;
    v8f acc = {};
    #pragma unroll
    for (int kk = 0; kk < CHUNK; kk += 32) {    // K-dim = t, 2 WMMAs
      const v8bf alo = *(const v8bf*)&VcT[irow][kk + 16 * h];       // A[i][t] = V[t][i]
      const v8bf ahi = *(const v8bf*)&VcT[irow][kk + 16 * h + 8];
      const v16bf a  = CAT16(alo, ahi);
      const v16bf b  = *(const v16bf*)&KcT[col][kk + 16 * h];       // B[t][j] = K[t][j]
      acc = WMMA_BF16(a, b, acc);
    }
    #pragma unroll
    for (int r = 0; r < 8; ++r)
      csum[(long)bc * D_ * D_ + (long)(wave * 16 + r + 8 * h) * D_ + col] = acc[r];
  }
}

// =====================================================================
// Kernel C: exclusive scan over chunk sums, seeded by initial state.
// Grid = B*D*D/256 = 128 blocks of 256.
// =====================================================================
__global__ __launch_bounds__(256)
void chunk_scan(const float* __restrict__ state,
                const float* __restrict__ csum,
                float* __restrict__ cstart) {
  const long e = (long)blockIdx.x * blockDim.x + threadIdx.x;
  if (e >= (long)B_ * D_ * D_) return;
  const int b  = (int)(e / (D_ * D_));
  const int ij = (int)(e % (D_ * D_));
  float run = state[e];
  for (int c = 0; c < NCH; ++c) {
    const long o = ((long)(b * NCH + c)) * D_ * D_ + ij;
    cstart[o] = run;
    run += csum[o];
  }
}

// =====================================================================
// Kernel D: in-chunk sequential cumsum (exact f32) writing state_seq + y.
// Grid = B*NCH = 128 blocks of 256. Each thread owns a 4x4 state sub-tile.
// K/Q/V chunks staged via global_load_async_to_lds_b128 (ASYNCcnt);
// state_seq written with non-temporal b128 stores (write-once stream).
// =====================================================================
__global__ __launch_bounds__(256)
void chunk_seq(const float* __restrict__ Kw,
               const float* __restrict__ Qw,
               const float* __restrict__ Vw,
               const float* __restrict__ cstart,
               float* __restrict__ y,
               float* __restrict__ state_seq) {
  __shared__ float Kc[CHUNK][D_], Qc[CHUNK][D_], Vc[CHUNK][D_];  // 48 KB
  __shared__ float yred[16][D_];                                 // 4 KB
  const int  tid     = threadIdx.x;
  const int  bc      = blockIdx.x;
  const long rowbase = (long)bc * CHUNK;       // == b*T + c*CHUNK
  const long gbase   = rowbase * D_;

#if defined(HAVE_ASYNC_LDS)
  {
    #pragma unroll
    for (int k = 0; k < 4; ++k) {
      const int eo = (tid + k * 256) * 4;      // float offset, 16B-aligned
      __builtin_amdgcn_global_load_async_to_lds_b128((gv4p)(Kw + gbase + eo), (lv4p)(&Kc[0][0] + eo), 0, 0);
      __builtin_amdgcn_global_load_async_to_lds_b128((gv4p)(Qw + gbase + eo), (lv4p)(&Qc[0][0] + eo), 0, 0);
      __builtin_amdgcn_global_load_async_to_lds_b128((gv4p)(Vw + gbase + eo), (lv4p)(&Vc[0][0] + eo), 0, 0);
    }
    __builtin_amdgcn_s_wait_asynccnt(0);
  }
#else
  for (int idx = tid; idx < CHUNK * D_; idx += 256) {
    Kc[idx / D_][idx % D_] = Kw[gbase + idx];
    Qc[idx / D_][idx % D_] = Qw[gbase + idx];
    Vc[idx / D_][idx % D_] = Vw[gbase + idx];
  }
#endif
  __syncthreads();

  const int jt = tid & 15, it = tid >> 4;
  const int jb = jt * 4,   ib = it * 4;

  float S[4][4];
  {
    const float* cs = cstart + (long)bc * D_ * D_;
    #pragma unroll
    for (int a = 0; a < 4; ++a)
      #pragma unroll
      for (int bb = 0; bb < 4; ++bb)
        S[a][bb] = cs[(ib + a) * D_ + jb + bb];
  }

  for (int t = 0; t < CHUNK; ++t) {
    float v[4], k[4], q[4];
    #pragma unroll
    for (int a = 0; a < 4; ++a) { v[a] = Vc[t][ib + a]; q[a] = Qc[t][ib + a]; k[a] = Kc[t][jb + a]; }

    // inclusive update: S += V (x) K
    #pragma unroll
    for (int a = 0; a < 4; ++a)
      #pragma unroll
      for (int bb = 0; bb < 4; ++bb)
        S[a][bb] = fmaf(v[a], k[bb], S[a][bb]);

    // state_seq[b, t, i, j]: bandwidth-critical 134 MB stream -> NT b128 stores
    float* so = state_seq + ((rowbase + t) * D_) * D_;
    #pragma unroll
    for (int a = 0; a < 4; ++a) {
      v4f st = {S[a][0], S[a][1], S[a][2], S[a][3]};
      __builtin_nontemporal_store(st, reinterpret_cast<v4f*>(so + (ib + a) * D_ + jb));
    }

    // partial y_j = sum_i Q_i * S_ij over this thread's 4 rows
    #pragma unroll
    for (int bb = 0; bb < 4; ++bb) {
      float py = q[0] * S[0][bb];
      py = fmaf(q[1], S[1][bb], py);
      py = fmaf(q[2], S[2][bb], py);
      py = fmaf(q[3], S[3][bb], py);
      yred[it][jb + bb] = py;
    }
    __syncthreads();
    if (tid < D_) {
      float acc = 0.f;
      #pragma unroll
      for (int g = 0; g < 16; ++g) acc += yred[g][tid];
      y[(rowbase + t) * D_ + tid] = acc;
    }
    __syncthreads();
  }
}

// =====================================================================
// Host launch
// =====================================================================
extern "C" void kernel_launch(void* const* d_in, const int* in_sizes, int n_in,
                              void* d_out, int out_size, void* d_ws, size_t ws_size,
                              hipStream_t stream) {
  const float* x     = (const float*)d_in[0];  // [B,T,F]
  const float* state = (const float*)d_in[1];  // [B,1,D*D]
  const float* Wk    = (const float*)d_in[2];  // [F,D]
  const float* Wq    = (const float*)d_in[3];
  const float* Wv    = (const float*)d_in[4];
  const float* gamma = (const float*)d_in[5];  // [F]
  const float* beta  = (const float*)d_in[6];

  float* y         = (float*)d_out;                       // [B,T,D]
  float* state_seq = (float*)d_out + (size_t)NROW * D_;   // [B,T,D,D]

  // workspace: K,Q,V (8192x64 f32) + chunk sums + chunk starts + bf16 Wt
  float*  Kw     = (float*)d_ws;
  float*  Qw     = Kw + (size_t)NROW * D_;
  float*  Vw     = Qw + (size_t)NROW * D_;
  float*  csum   = Vw + (size_t)NROW * D_;
  float*  cstart = csum + (size_t)B_ * NCH * D_ * D_;
  bf16_t* Wt     = (bf16_t*)(cstart + (size_t)B_ * NCH * D_ * D_);  // [3][D][F]

  w_prep<<<(3 * F_ * D_ + 255) / 256, 256, 0, stream>>>(Wk, Wq, Wv, Wt);
  proj_ln_gemm<<<NROW / 16, 32, 0, stream>>>(x, Wt, gamma, beta, Kw, Qw, Vw);
  chunk_outer_sum<<<B_ * NCH, 128, 0, stream>>>(Kw, Vw, csum);
  chunk_scan<<<(B_ * D_ * D_) / 256, 256, 0, stream>>>(state, csum, cstart);
  chunk_seq<<<B_ * NCH, 256, 0, stream>>>(Kw, Qw, Vw, cstart, y, state_seq);
}
